// GroupedQueryAttention_28501402976336
// MI455X (gfx1250) — compile-verified, matrix operand order checked by reading, not run on hardware
//
#include <hip/hip_runtime.h>
#include <hip/hip_bf16.h>
#include <cstdint>

typedef __attribute__((ext_vector_type(16))) _Float16     v16h;
typedef __attribute__((ext_vector_type(4)))  _Float16     v4h;
typedef __attribute__((ext_vector_type(8)))  float        v8f;
typedef __attribute__((ext_vector_type(4)))  unsigned int v4u;
typedef __attribute__((ext_vector_type(8)))  int          v8i;
typedef __attribute__((ext_vector_type(4)))  int          v4i;

#define WMMA_F16(a, b, c) \
  __builtin_amdgcn_wmma_f32_16x16x32_f16(false, (a), false, (b), (short)0, (c), false, false)

static constexpr int   Bb  = 4;
static constexpr int   Tt  = 2048;
static constexpr int   Dd  = 1024;
static constexpr int   Hh  = 16;
static constexpr int   KVH = 4;
static constexpr int   HD  = 64;
static constexpr int   GROUPS = Hh / KVH;  // 4
static constexpr float EPSf = 1e-6f;

// ---------------------------------------------------------------------------
// WMMA A/B fragment loader from f16 memory (global or generic/LDS).
// A 16x32 (ISA 7.12.2): lanes 0-15 = rows 0-15, K={0..7,16..23};
// lanes 16-31 = rows 0-15, K={8..15,24..31}. B 32x16 mirrored, lane = column
// (column data contiguous when source is row-major with that column as a row).
// Two contiguous 16B groups per lane -> 2x b128 loads, no VALU conversion.
// ---------------------------------------------------------------------------
__device__ __forceinline__ v16h ldA_f16(const _Float16* __restrict__ base, int ld,
                                        int row0, int k0, int lane) {
  const int r  = lane & 15;
  const int ks = (lane >> 4) << 3;
  const _Float16* p = base + (size_t)(row0 + r) * ld + (k0 + ks);
  v16h a;
#pragma unroll
  for (int i = 0; i < 8; ++i) a[i] = p[i];
#pragma unroll
  for (int i = 0; i < 8; ++i) a[8 + i] = p[16 + i];
  return a;
}

// ---------------------------------------------------------------------------
// TDM: 2D tile load Global -> LDS (f16 elements). Tile = rows x cols, tensor
// row stride = stride_e elements. Descriptor per CDNA5 ISA.
// ---------------------------------------------------------------------------
__device__ __forceinline__ void tdm_load_2d_f16(const _Float16* gsrc,
                                                uint32_t lds_byte_addr,
                                                int rows, int cols, int stride_e,
                                                int tensor_rows) {
  const uint64_t ga = (uint64_t)(uintptr_t)gsrc;
  v4u g0;
  g0[0] = 1u;                                    // count=1, user descriptor
  g0[1] = lds_byte_addr;                         // lds_addr (bytes)
  g0[2] = (uint32_t)(ga & 0xffffffffu);          // global_addr[31:0]
  g0[3] = (uint32_t)((ga >> 32) & 0x01ffffffu)   // global_addr[56:32]
          | (2u << 30);                          // type = 2 (image)
  v8i g1;
  g1[0] = (int)(1u << 16);                               // wg_mask=0, data_size=2B
  g1[1] = (int)(((uint32_t)cols & 0xffffu) << 16);       // tensor_dim0[15:0]
  g1[2] = (int)((((uint32_t)cols >> 16) & 0xffffu)       // tensor_dim0[31:16]
          | (((uint32_t)tensor_rows & 0xffffu) << 16));  // tensor_dim1[15:0]
  g1[3] = (int)(((((uint32_t)tensor_rows >> 16)) & 0xffffu)
          | (((uint32_t)cols & 0xffffu) << 16));         // tile_dim0 = cols
  g1[4] = rows & 0xffff;                                 // tile_dim1, tile_dim2=0
  g1[5] = stride_e;                                      // tensor_dim0_stride lo32
  g1[6] = 0;
  g1[7] = 0;
  const v4i z4 = {0, 0, 0, 0};
#if __clang_major__ >= 23
  const v8i z8 = {0, 0, 0, 0, 0, 0, 0, 0};
  __builtin_amdgcn_tensor_load_to_lds(g0, g1, z4, z4, z8, 0);
#else
  __builtin_amdgcn_tensor_load_to_lds(g0, g1, z4, z4, 0);
#endif
}

// ---------------------------------------------------------------------------
// Kernel 0: one-time f32 -> f16 conversion (x and weights), vectorized x4.
// ---------------------------------------------------------------------------
__global__ __launch_bounds__(256)
void cvt_f32_f16_kernel(const float* __restrict__ in, _Float16* __restrict__ out,
                        int n) {
  const int i = (blockIdx.x * 256 + threadIdx.x) * 4;
  if (i + 3 < n) {
    const float4 v = *(const float4*)(in + i);
    v4h h;
    h[0] = (_Float16)v.x; h[1] = (_Float16)v.y;
    h[2] = (_Float16)v.z; h[3] = (_Float16)v.w;
    *(v4h*)(out + i) = h;
  }
}

// ---------------------------------------------------------------------------
// Kernel 1: projection + per-head RMSNorm + RoPE. Block = 4 waves; the 16x1024
// f16 activation row-block is TDM-staged into LDS once and shared by all
// waves; each wave owns one head (16x64 tile, 4 WMMA accumulators over K=D).
// Norm/RoPE epilogue is register-resident (half-wave shuffles; RoPE pairs
// (j, j+16) = accumulators t=0/t=1 in the same lane). outp: [b][NH][t][HD].
// ---------------------------------------------------------------------------
__global__ __launch_bounds__(128)
void qkv_proj_kernel(const _Float16* __restrict__ x16, const _Float16* __restrict__ W16,
                     const float* __restrict__ hscale,
                     const float* __restrict__ cosb, const float* __restrict__ sinb,
                     _Float16* __restrict__ outp, int NH, int do_norm_rope) {
  const int m0   = blockIdx.x * 16;               // token-row tile (over B*T)
  const int wave = threadIdx.x >> 5;
  const int lane = threadIdx.x & 31;
  const int head = blockIdx.y * 4 + wave;         // q: gridDim.y=4 ; k/v: =1
  const int n0   = head * HD;

  __shared__ _Float16 xs[16][Dd];                 // 32 KB staged A block
  if (wave == 0) {
    tdm_load_2d_f16(x16 + (size_t)m0 * Dd, (uint32_t)(uintptr_t)&xs[0][0],
                    16, Dd, Dd, Bb * Tt);
    __builtin_amdgcn_s_wait_tensorcnt((short)0);
  }
  __syncthreads();

  const int hiOff = (lane >> 4) << 3;
  const int nlo   = lane & 15;

  v8f c[4] = {};
  for (int k0 = 0; k0 < Dd; k0 += 32) {
    v16h a;
    {
      const _Float16* ap = &xs[nlo][k0 + hiOff];
#pragma unroll
      for (int i = 0; i < 8; ++i) a[i] = ap[i];
#pragma unroll
      for (int i = 0; i < 8; ++i) a[8 + i] = ap[16 + i];
    }
#pragma unroll
    for (int t = 0; t < 4; ++t) {
      // W16 row-major (N,K); y = x W^T -> B[k][n] = W16[n0+16t+n][k]
      const v16h b = ldA_f16(W16, Dd, n0 + 16 * t, k0, lane);
      c[t] = WMMA_F16(a, b, c[t]);
    }
  }

  float rn[8];
  float hs[4] = {1.f, 1.f, 1.f, 1.f};
  if (do_norm_rope) {
#pragma unroll
    for (int t = 0; t < 4; ++t) hs[t] = hscale[nlo + 16 * t];
#pragma unroll
    for (int r = 0; r < 8; ++r) {
      float s = 0.f;
#pragma unroll
      for (int t = 0; t < 4; ++t) s += c[t][r] * c[t][r];
#pragma unroll
      for (int off = 1; off < 16; off <<= 1) s += __shfl_xor(s, off, 32);
      rn[r] = rsqrtf(s * (1.0f / HD) + EPSf);
    }
  } else {
#pragma unroll
    for (int r = 0; r < 8; ++r) rn[r] = 1.f;
  }

#pragma unroll
  for (int r = 0; r < 8; ++r) {
    const int row = m0 + r + hiOff;
    const int bb  = row / Tt;
    const int tt  = row - bb * Tt;
    float v0 = c[0][r], v1 = c[1][r], v2 = c[2][r], v3 = c[3][r];
    if (do_norm_rope) {
      v0 *= rn[r] * hs[0]; v1 *= rn[r] * hs[1];
      v2 *= rn[r] * hs[2]; v3 *= rn[r] * hs[3];
      const float cs = cosb[tt * 16 + nlo];
      const float sn = sinb[tt * 16 + nlo];
      const float r0 = v0 * cs - v1 * sn;   // dims [0,16)
      const float r1 = v0 * sn + v1 * cs;   // dims [16,32)
      v0 = r0; v1 = r1;                     // dims >= 32 pass through
    }
    _Float16* o = outp + (((size_t)bb * NH + head) * Tt + tt) * HD;
    o[nlo]      = (_Float16)v0;
    o[16 + nlo] = (_Float16)v1;
    o[32 + nlo] = (_Float16)v2;
    o[48 + nlo] = (_Float16)v3;
  }
}

// ---------------------------------------------------------------------------
// Kernel 2: causal flash attention. Block = 4 waves = 64 query rows of one
// (b,h). K/V 32x64 f16 tiles are double-buffered in LDS via the Tensor Data
// Mover: wave 0 issues the next tile's DMAs before waiting TENSORcnt<=2 on
// the current one (TDM ops complete in order), overlapping DMA with the
// 8 WMMAs + online softmax per tile.
// ---------------------------------------------------------------------------
__global__ __launch_bounds__(128)
void flash_attn_kernel(const _Float16* __restrict__ qb, const _Float16* __restrict__ kb,
                       const _Float16* __restrict__ vb, _Float16* __restrict__ yb) {
  const int qt0  = blockIdx.x * 64;
  const int h    = blockIdx.y;
  const int b    = blockIdx.z;
  const int kvh  = h / GROUPS;
  const int wave = threadIdx.x >> 5;
  const int lane = threadIdx.x & 31;
  const int qt   = qt0 + wave * 16;

  const _Float16* Q  = qb + ((size_t)b * Hh  + h)   * Tt * HD;
  const _Float16* Kp = kb + ((size_t)b * KVH + kvh) * Tt * HD;
  const _Float16* Vp = vb + ((size_t)b * KVH + kvh) * Tt * HD;

  __shared__ _Float16 ktile[2][32][HD];   // 8 KB double-buffered DMA target
  __shared__ _Float16 vtile[2][32][HD];   // 8 KB
  __shared__ _Float16 plds[4][16][40];    // per-wave P transpose staging

  const v16h qa0 = ldA_f16(Q, HD, qt, 0,  lane);
  const v16h qa1 = ldA_f16(Q, HD, qt, 32, lane);

  v8f   o[4] = {};
  float m[8], l[8];
#pragma unroll
  for (int r = 0; r < 8; ++r) { m[r] = -1e30f; l[r] = 0.f; }

  const int hiOff = (lane >> 4) << 3;
  const int nlo   = lane & 15;

  const int nTiles = (qt0 + 64) >> 5;
  if (wave == 0) {
    tdm_load_2d_f16(Kp, (uint32_t)(uintptr_t)&ktile[0][0][0], 32, HD, HD, Tt);
    tdm_load_2d_f16(Vp, (uint32_t)(uintptr_t)&vtile[0][0][0], 32, HD, HD, Tt);
  }

  for (int it = 0; it < nTiles; ++it) {
    const int jt  = it * 32;
    const int cur = it & 1;
    if (wave == 0) {
      if (it + 1 < nTiles) {
        const int nxt = (it + 1) & 1;
        tdm_load_2d_f16(Kp + (size_t)(jt + 32) * HD,
                        (uint32_t)(uintptr_t)&ktile[nxt][0][0], 32, HD, HD, Tt);
        tdm_load_2d_f16(Vp + (size_t)(jt + 32) * HD,
                        (uint32_t)(uintptr_t)&vtile[nxt][0][0], 32, HD, HD, Tt);
        __builtin_amdgcn_s_wait_tensorcnt((short)2);  // current tile done
      } else {
        __builtin_amdgcn_s_wait_tensorcnt((short)0);
      }
    }
    __syncthreads();

    // ---- S = Q K^T : keys jt..jt+31 as two 16-col tiles, HD split 2x32 ----
    v8f s0 = {}, s1 = {};
    {
      v16h kf;
#pragma unroll
      for (int half = 0; half < 2; ++half) {
        const int d0 = half * 32;
        const _Float16* kp0 = &ktile[cur][nlo][d0 + hiOff];
#pragma unroll
        for (int i = 0; i < 8; ++i) kf[i] = kp0[i];
#pragma unroll
        for (int i = 0; i < 8; ++i) kf[8 + i] = kp0[16 + i];
        s0 = WMMA_F16(half ? qa1 : qa0, kf, s0);
        const _Float16* kp1 = &ktile[cur][16 + nlo][d0 + hiOff];
#pragma unroll
        for (int i = 0; i < 8; ++i) kf[i] = kp1[i];
#pragma unroll
        for (int i = 0; i < 8; ++i) kf[8 + i] = kp1[16 + i];
        s1 = WMMA_F16(half ? qa1 : qa0, kf, s1);
      }
    }

    // ---- online softmax; row r+hiOff lives entirely in this half-wave ----
#pragma unroll
    for (int r = 0; r < 8; ++r) {
      const int row = qt + r + hiOff;
      float e0 = s0[r] * 0.125f;          // 1/sqrt(64)
      float e1 = s1[r] * 0.125f;
      if (jt + nlo > row)      e0 = -1e30f;
      if (jt + 16 + nlo > row) e1 = -1e30f;
      float mx = fmaxf(e0, e1);
#pragma unroll
      for (int off = 1; off < 16; off <<= 1) mx = fmaxf(mx, __shfl_xor(mx, off, 32));
      const float mnew  = fmaxf(m[r], mx);
      const float alpha = __expf(m[r] - mnew);
      const float p0 = __expf(e0 - mnew);
      const float p1 = __expf(e1 - mnew);
      float rs = p0 + p1;
#pragma unroll
      for (int off = 1; off < 16; off <<= 1) rs += __shfl_xor(rs, off, 32);
      l[r] = l[r] * alpha + rs;
      m[r] = mnew;
#pragma unroll
      for (int t2 = 0; t2 < 4; ++t2) o[t2][r] *= alpha;
      plds[wave][r + hiOff][nlo]      = (_Float16)p0;
      plds[wave][r + hiOff][16 + nlo] = (_Float16)p1;
    }

    // ---- P (16x32) as A-fragment from per-wave LDS (DS in-order in wave) --
    v16h pa;
#pragma unroll
    for (int i = 0; i < 8; ++i) pa[i]     = plds[wave][nlo][hiOff + i];
#pragma unroll
    for (int i = 0; i < 8; ++i) pa[8 + i] = plds[wave][nlo][16 + hiOff + i];

    // ---- O += P V : V tile 32 keys x 64 dims, 4 N-subtiles ----------------
#pragma unroll
    for (int t2 = 0; t2 < 4; ++t2) {
      v16h vf;
      const int col = t2 * 16 + nlo;
#pragma unroll
      for (int i = 0; i < 8; ++i) vf[i]     = vtile[cur][hiOff + i][col];
#pragma unroll
      for (int i = 0; i < 8; ++i) vf[8 + i] = vtile[cur][16 + hiOff + i][col];
      o[t2] = WMMA_F16(pa, vf, o[t2]);
    }
    __syncthreads();   // all waves done with buffer `cur` before it is reused
  }

  // ---- epilogue: normalize, write y in [b][t][h][hd] (== [B*T][D]) f16 ----
#pragma unroll
  for (int t2 = 0; t2 < 4; ++t2) {
#pragma unroll
    for (int r = 0; r < 8; ++r) {
      const int row  = qt + r + hiOff;
      const int dcol = t2 * 16 + nlo;
      const float val = o[t2][r] / l[r];
      yb[(((size_t)b * Tt + row) * Hh + h) * HD + dcol] = (_Float16)val;
    }
  }
}

// ---------------------------------------------------------------------------
// Kernel 3: out = Y @ Wo^T. Block = 4 waves sharing a TDM-staged 16x1024 Y
// row-block in LDS; wave w computes 16x64 output columns (4 WMMA accums).
// ---------------------------------------------------------------------------
__global__ __launch_bounds__(128)
void out_proj_kernel(const _Float16* __restrict__ yb, const _Float16* __restrict__ Wo16,
                     float* __restrict__ out) {
  const int m0   = blockIdx.x * 16;
  const int wave = threadIdx.x >> 5;
  const int lane = threadIdx.x & 31;
  const int n0   = blockIdx.y * 256 + wave * 64;

  __shared__ _Float16 ys[16][Dd];
  if (wave == 0) {
    tdm_load_2d_f16(yb + (size_t)m0 * Dd, (uint32_t)(uintptr_t)&ys[0][0],
                    16, Dd, Dd, Bb * Tt);
    __builtin_amdgcn_s_wait_tensorcnt((short)0);
  }
  __syncthreads();

  const int hiOff = (lane >> 4) << 3;
  const int nlo   = lane & 15;

  v8f c[4] = {};
  for (int k0 = 0; k0 < Dd; k0 += 32) {
    v16h a;
    {
      const _Float16* ap = &ys[nlo][k0 + hiOff];
#pragma unroll
      for (int i = 0; i < 8; ++i) a[i] = ap[i];
#pragma unroll
      for (int i = 0; i < 8; ++i) a[8 + i] = ap[16 + i];
    }
#pragma unroll
    for (int t = 0; t < 4; ++t) {
      const v16h b = ldA_f16(Wo16, Dd, n0 + 16 * t, k0, lane);
      c[t] = WMMA_F16(a, b, c[t]);
    }
  }

  const int mr = hiOff;
#pragma unroll
  for (int t = 0; t < 4; ++t)
#pragma unroll
    for (int r = 0; r < 8; ++r)
      out[(size_t)(m0 + mr + r) * Dd + n0 + 16 * t + nlo] = c[t][r];
}

// ---------------------------------------------------------------------------
// Launcher. Inputs (all f32): 0:x 1:cos 2:sin 3:Wq 4:Wk 5:Wv 6:Wo 7:q_s 8:k_s
// Workspace (f16): x16(16MB) Wq16(2MB) Wk16/Wv16(.5MB) Wo16(2MB)
//                  qbuf(16MB) kbuf/vbuf(4MB) ybuf(16MB)  ~= 61MB total.
// ---------------------------------------------------------------------------
extern "C" void kernel_launch(void* const* d_in, const int* in_sizes, int n_in,
                              void* d_out, int out_size, void* d_ws, size_t ws_size,
                              hipStream_t stream) {
  (void)in_sizes; (void)n_in; (void)out_size; (void)ws_size;
  const float* x    = (const float*)d_in[0];
  const float* cosb = (const float*)d_in[1];
  const float* sinb = (const float*)d_in[2];
  const float* Wq   = (const float*)d_in[3];
  const float* Wk   = (const float*)d_in[4];
  const float* Wv   = (const float*)d_in[5];
  const float* Wo   = (const float*)d_in[6];
  const float* qs   = (const float*)d_in[7];
  const float* ks   = (const float*)d_in[8];

  const size_t nX  = (size_t)Bb * Tt * Dd;      // 8388608
  const size_t nWq = (size_t)Dd * Dd;           // 1048576
  const size_t nWk = (size_t)KVH * HD * Dd;     // 262144

  _Float16* x16  = (_Float16*)d_ws;
  _Float16* Wq16 = x16  + nX;
  _Float16* Wk16 = Wq16 + nWq;
  _Float16* Wv16 = Wk16 + nWk;
  _Float16* Wo16 = Wv16 + nWk;
  _Float16* qbuf = Wo16 + nWq;                  // B*H*T*HD
  _Float16* kbuf = qbuf + (size_t)Bb * Hh  * Tt * HD;
  _Float16* vbuf = kbuf + (size_t)Bb * KVH * Tt * HD;
  _Float16* ybuf = vbuf + (size_t)Bb * KVH * Tt * HD;

  const int M = Bb * Tt;  // 8192 token rows

  // ---- one-time f32 -> f16 conversions ----
  cvt_f32_f16_kernel<<<dim3((unsigned)(nX  / 1024)), dim3(256), 0, stream>>>(x,  x16,  (int)nX);
  cvt_f32_f16_kernel<<<dim3((unsigned)(nWq / 1024)), dim3(256), 0, stream>>>(Wq, Wq16, (int)nWq);
  cvt_f32_f16_kernel<<<dim3((unsigned)(nWk / 1024)), dim3(256), 0, stream>>>(Wk, Wk16, (int)nWk);
  cvt_f32_f16_kernel<<<dim3((unsigned)(nWk / 1024)), dim3(256), 0, stream>>>(Wv, Wv16, (int)nWk);
  cvt_f32_f16_kernel<<<dim3((unsigned)(nWq / 1024)), dim3(256), 0, stream>>>(Wo, Wo16, (int)nWq);

  // ---- fused projection + norm + rope ----
  qkv_proj_kernel<<<dim3(M / 16, Hh / 4), dim3(128), 0, stream>>>(x16, Wq16, qs, cosb, sinb, qbuf, Hh,  1);
  qkv_proj_kernel<<<dim3(M / 16, 1),      dim3(128), 0, stream>>>(x16, Wk16, ks, cosb, sinb, kbuf, KVH, 1);
  qkv_proj_kernel<<<dim3(M / 16, 1),      dim3(128), 0, stream>>>(x16, Wv16, ks, cosb, sinb, vbuf, KVH, 0);

  // ---- causal flash attention (GQA) ----
  flash_attn_kernel<<<dim3(Tt / 64, Hh, Bb), dim3(128), 0, stream>>>(qbuf, kbuf, vbuf, ybuf);

  // ---- output projection ----
  out_proj_kernel<<<dim3(M / 16, Dd / 256), dim3(128), 0, stream>>>(ybuf, Wo16, (float*)d_out);
}
